// S5Operator_68109591380651
// MI455X (gfx1250) — matrix operand order repaired and checked by reading, not compiled, not run on hardware
//
#include <hip/hip_runtime.h>
#include <hip/hip_bf16.h>
#include <math.h>

// ---------------- problem constants ----------------
#define BSZ   8
#define LSEQ  4096
#define HDIM  512           // features (also GEMM K for phase 1, N for phase 2)
#define PDIM  256           // complex state dim; packed real width = 2*PDIM = 512
#define MROWS (BSZ * LSEQ)  // 32768
#define KFULL 512           // both GEMMs contract over 512
#define NFULL 512           // both GEMMs produce 512 cols

#define NCHUNK 64           // scan chunks over L
#define CLEN   64           // chunk length (NCHUNK*CLEN == LSEQ)

typedef __attribute__((ext_vector_type(16))) __bf16 v16bf;
typedef __attribute__((ext_vector_type(8)))  float  v8f;
typedef __attribute__((ext_vector_type(4)))  float  f32x4;

// =====================================================================
// 0) Discretization: lam_bar = exp(Lambda*step), coef = (lam_bar-1)/Lambda
//    lamcoef layout: [0..255]=lam_re [256..511]=lam_im [512..767]=coef_re [768..1023]=coef_im
// =====================================================================
__global__ void s5_precompute(const float* __restrict__ Lre,
                              const float* __restrict__ Lim,
                              const float* __restrict__ log_step,
                              float* __restrict__ lamcoef) {
  int p = threadIdx.x;
  if (p >= PDIM) return;
  float lr = Lre[p], li = Lim[p];
  float step = __expf(log_step[p]);
  float mag = __expf(lr * step);
  float ang = li * step;
  float lam_re = mag * __cosf(ang);
  float lam_im = mag * __sinf(ang);
  // (lam-1)/Lambda  via complex division
  float a = lam_re - 1.0f, b = lam_im;
  float den = lr * lr + li * li;
  float inv = 1.0f / den;
  float c_re = (a * lr + b * li) * inv;
  float c_im = (b * lr - a * li) * inv;
  lamcoef[p]             = lam_re;
  lamcoef[PDIM + p]      = lam_im;
  lamcoef[2 * PDIM + p]  = c_re;
  lamcoef[3 * PDIM + p]  = c_im;
}

// =====================================================================
// 1) Pack weight matrices into per-lane WMMA B-operand fragments (bf16).
//    B-fragment layout (16x16x32 bf16): lane n = lane&15, K half = lane>>4,
//    element e of v16bf holds K = kt*32 + (lane>=16?16:0) + e.
//    Fragment storage: ((kt*32 + nt)*32 + lane) * 16 bf16, 32B aligned.
//    mode 0: W1[k=h][n] = n<256 ?  Re(coef[n]*B_tilde[n,k]) : Im(coef[n-256]*B_tilde[n-256,k])
//    mode 1: W2[k][n=h] = k<256 ?  2*C_re[n,k]              : -2*C_im[n,k-256]
// =====================================================================
__global__ void s5_pack_weights(const float* __restrict__ Bmat,   // (P,H,2)
                                const float* __restrict__ Cmat,   // (H,P,2)
                                const float* __restrict__ lamcoef,
                                __bf16* __restrict__ Wp,
                                int mode) {
  int tid  = blockIdx.x * blockDim.x + threadIdx.x;   // 16384 threads
  int lane = tid & 31;
  int tile = tid >> 5;                                // kt*32 + nt
  int kt = tile >> 5;
  int nt = tile & 31;
  int kb = kt * 32 + ((lane >> 4) << 4);
  int n  = nt * 16 + (lane & 15);

  v16bf frag;
#pragma unroll
  for (int e = 0; e < 16; ++e) {
    int k = kb + e;
    float val;
    if (mode == 0) {
      if (n < PDIM) {
        int p = n;
        float cr = lamcoef[2 * PDIM + p], ci = lamcoef[3 * PDIM + p];
        float br = Bmat[(p * HDIM + k) * 2 + 0];
        float bi = Bmat[(p * HDIM + k) * 2 + 1];
        val = cr * br - ci * bi;                 // Re(B_bar)
      } else {
        int p = n - PDIM;
        float cr = lamcoef[2 * PDIM + p], ci = lamcoef[3 * PDIM + p];
        float br = Bmat[(p * HDIM + k) * 2 + 0];
        float bi = Bmat[(p * HDIM + k) * 2 + 1];
        val = cr * bi + ci * br;                 // Im(B_bar)
      }
    } else {
      if (k < PDIM) {
        val =  2.0f * Cmat[(n * PDIM + k) * 2 + 0];          //  2*C_re
      } else {
        val = -2.0f * Cmat[(n * PDIM + (k - PDIM)) * 2 + 1]; // -2*C_im
      }
    }
    frag[e] = (__bf16)val;
  }
  *(v16bf*)(Wp + (size_t)tid * 16) = frag;
}

// =====================================================================
// 2) GEMM: Cout(M,512) = A(M,512 f32 -> bf16) x Wp(packed bf16 frags),
//    f32 accumulate via v_wmma_f32_16x16x32_bf16.
//    Block = 256 threads = 8 waves (2 in M x 4 in N); block tile 32M x 512N.
//    B-fragment loads for a K-step are batched into 8 distinct registers
//    *before* the 8-WMMA burst so the load clause overlaps WMMA issue,
//    and the K+2 weight fragments are prefetched (global_prefetch_b8).
//    fuse!=0: epilogue  y = acc + D[n]*U[m,n]; gelu_tanh(y)  (phase-2 only)
// =====================================================================
__global__ __launch_bounds__(256)
void s5_gemm_wmma(const float* __restrict__ A,
                  const __bf16* __restrict__ Wp,
                  float* __restrict__ Cout,
                  const float* __restrict__ U,
                  const float* __restrict__ Dvec,
                  int fuse) {
  const int lane = threadIdx.x & 31;
  const int wave = threadIdx.x >> 5;
  const int m0 = blockIdx.x * 32 + (wave & 1) * 16;
  const int n0 = (wave >> 1) * 128;
  const int nt0 = n0 >> 4;

  const int rowA = m0 + (lane & 15);
  const int kb   = (lane >> 4) ? 16 : 0;      // A layout: half-wave K offset
  const float* arow = A + (size_t)rowA * KFULL;

  v8f acc[8];
#pragma unroll
  for (int t = 0; t < 8; ++t) acc[t] = (v8f)0.0f;

#pragma unroll 2
  for (int kt = 0; kt < KFULL / 32; ++kt) {
    const int k0 = kt * 32;

    // ---- batch-issue all 8 B fragments for this K-step (one clause)
    v16bf bfr[8];
#pragma unroll
    for (int t = 0; t < 8; ++t) {
      bfr[t] = *(const v16bf*)(Wp + (size_t)((kt * 32 + nt0 + t) * 32 + lane) * 16);
    }

    // ---- prefetch weight fragments two K-steps ahead
    if (kt + 2 < KFULL / 32) {
      __builtin_prefetch(Wp + (size_t)(((kt + 2) * 32 + nt0) * 32 + lane) * 16, 0, 3);
    }

    // ---- A fragment: lane holds K = k0+kb+{0..15}
    f32x4 x0 = *(const f32x4*)(arow + k0 + kb + 0);
    f32x4 x1 = *(const f32x4*)(arow + k0 + kb + 4);
    f32x4 x2 = *(const f32x4*)(arow + k0 + kb + 8);
    f32x4 x3 = *(const f32x4*)(arow + k0 + kb + 12);
    v16bf a;
#pragma unroll
    for (int e = 0; e < 4; ++e) {
      a[e]      = (__bf16)x0[e];
      a[4 + e]  = (__bf16)x1[e];
      a[8 + e]  = (__bf16)x2[e];
      a[12 + e] = (__bf16)x3[e];
    }

    // ---- 8 WMMAs; B loads above drain while the matrix pipe runs
#pragma unroll
    for (int t = 0; t < 8; ++t) {
      acc[t] = __builtin_amdgcn_wmma_f32_16x16x32_bf16(
          false, a, false, bfr[t], (short)0, acc[t], false, false);
    }
  }

  // ---- writeout (C/D layout: VGPR v -> M = v + (lane>=16 ? 8:0), N = lane&15)
  const int mlane = m0 + ((lane >> 4) << 3);
#pragma unroll
  for (int t = 0; t < 8; ++t) {
    int n = n0 + t * 16 + (lane & 15);
#pragma unroll
    for (int v = 0; v < 8; ++v) {
      int m = mlane + v;
      float y = acc[t][v];
      if (fuse) {
        y += Dvec[n] * U[(size_t)m * HDIM + n];
        // tanh-approx GELU (jax.nn.gelu default)
        float y3 = y * y * y;
        y = 0.5f * y * (1.0f + tanhf(0.7978845608028654f * (y + 0.044715f * y3)));
      }
      Cout[(size_t)m * NFULL + n] = y;
    }
  }
}

// =====================================================================
// 3) Scan pass 1: per-chunk local inclusive scan, in place over Bu.
//    block = 256 threads (one per p); grid = BSZ*NCHUNK. All dword
//    accesses coalesced (consecutive p across lanes).
// =====================================================================
__global__ __launch_bounds__(256)
void s5_scan_local(float* __restrict__ Bu,
                   const float* __restrict__ lamcoef,
                   float* __restrict__ carry) {
  int b = blockIdx.x / NCHUNK;
  int c = blockIdx.x % NCHUNK;
  int p = threadIdx.x;
  float lr = lamcoef[p], li = lamcoef[PDIM + p];
  float xr = 0.0f, xi = 0.0f;
  size_t idx = ((size_t)(b * LSEQ + c * CLEN)) * NFULL + p;
#pragma unroll 4
  for (int i = 0; i < CLEN; ++i) {
    float br = Bu[idx];
    float bi = Bu[idx + PDIM];
    float nr = fmaf(lr, xr, fmaf(-li, xi, br));
    float ni = fmaf(lr, xi, fmaf( li, xr, bi));
    xr = nr; xi = ni;
    Bu[idx]        = xr;
    Bu[idx + PDIM] = xi;
    idx += NFULL;
  }
  size_t cidx = ((size_t)(b * NCHUNK + c) * PDIM + p) * 2;
  carry[cidx]     = xr;
  carry[cidx + 1] = xi;
}

// =====================================================================
// 4) Scan pass 2: serial scan over chunk aggregates -> exclusive carries.
//    grid = BSZ blocks x 256 threads (one thread per (b,p)).
// =====================================================================
__global__ void s5_scan_carry(float* __restrict__ carry,
                              const float* __restrict__ lamcoef) {
  int b = blockIdx.x;
  int p = threadIdx.x;
  float lr = lamcoef[p], li = lamcoef[PDIM + p];
  // lam^CLEN via 6 complex squarings (CLEN = 64 = 2^6)
  float ar = lr, ai = li;
#pragma unroll
  for (int s = 0; s < 6; ++s) {
    float nr = ar * ar - ai * ai;
    float ni = 2.0f * ar * ai;
    ar = nr; ai = ni;
  }
  float er = 0.0f, ei = 0.0f;   // exclusive prefix state
  for (int c = 0; c < NCHUNK; ++c) {
    size_t cidx = ((size_t)(b * NCHUNK + c) * PDIM + p) * 2;
    float gr = carry[cidx], gi = carry[cidx + 1];
    carry[cidx]     = er;
    carry[cidx + 1] = ei;
    float nr = fmaf(ar, er, fmaf(-ai, ei, gr));
    float ni = fmaf(ar, ei, fmaf( ai, er, gi));
    er = nr; ei = ni;
  }
}

// =====================================================================
// 5) Scan pass 3: xs[l0+i] += lam^(i+1) * carry_excl   (coalesced fixup)
// =====================================================================
__global__ __launch_bounds__(256)
void s5_scan_fixup(float* __restrict__ Bu,
                   const float* __restrict__ lamcoef,
                   const float* __restrict__ carry) {
  int b = blockIdx.x / NCHUNK;
  int c = blockIdx.x % NCHUNK;
  int p = threadIdx.x;
  float lr = lamcoef[p], li = lamcoef[PDIM + p];
  size_t cidx = ((size_t)(b * NCHUNK + c) * PDIM + p) * 2;
  float yr = carry[cidx], yi = carry[cidx + 1];
  size_t idx = ((size_t)(b * LSEQ + c * CLEN)) * NFULL + p;
#pragma unroll 4
  for (int i = 0; i < CLEN; ++i) {
    float nr = lr * yr - li * yi;   // lam^(i+1) * carry
    float ni = lr * yi + li * yr;
    yr = nr; yi = ni;
    Bu[idx]        += yr;
    Bu[idx + PDIM] += yi;
    idx += NFULL;
  }
}

// =====================================================================
// host-side launcher
// =====================================================================
extern "C" void kernel_launch(void* const* d_in, const int* in_sizes, int n_in,
                              void* d_out, int out_size, void* d_ws, size_t ws_size,
                              hipStream_t stream) {
  const float* u        = (const float*)d_in[0];   // (B,L,H)
  const float* Lre      = (const float*)d_in[1];   // (P,)
  const float* Lim      = (const float*)d_in[2];   // (P,)
  const float* Bmat     = (const float*)d_in[3];   // (P,H,2)
  const float* Cmat     = (const float*)d_in[4];   // (H,P,2)
  const float* Dvec     = (const float*)d_in[5];   // (H,)
  const float* log_step = (const float*)d_in[6];   // (P,1)
  float* out = (float*)d_out;

  // workspace layout
  char* w = (char*)d_ws;
  float*  Bu      = (float*)w;                               // 32768*512*4 = 64 MB
  size_t  off     = (size_t)MROWS * NFULL * sizeof(float);
  __bf16* W1p     = (__bf16*)(w + off); off += (size_t)KFULL * NFULL * sizeof(__bf16); // 512 KB
  __bf16* W2p     = (__bf16*)(w + off); off += (size_t)KFULL * NFULL * sizeof(__bf16); // 512 KB
  float*  lamcoef = (float*)(w + off);  off += 4 * PDIM * sizeof(float);               // 4 KB
  float*  carry   = (float*)(w + off);  // 8*64*256*2*4 = 1 MB

  // 0) discretization
  s5_precompute<<<1, PDIM, 0, stream>>>(Lre, Lim, log_step, lamcoef);

  // 1) pack weights into WMMA fragment layout (16 Ktiles * 32 Ntiles * 32 lanes)
  {
    int threads = (KFULL / 32) * (NFULL / 16) * 32;   // 16384
    s5_pack_weights<<<threads / 256, 256, 0, stream>>>(Bmat, Cmat, lamcoef, W1p, 0);
    s5_pack_weights<<<threads / 256, 256, 0, stream>>>(Bmat, Cmat, lamcoef, W2p, 1);
  }

  // 2) GEMM1: Bu = u @ [Re(B_bar)^T | Im(B_bar)^T]
  s5_gemm_wmma<<<MROWS / 32, 256, 0, stream>>>(u, W1p, Bu, nullptr, nullptr, 0);

  // 3-5) chunked associative scan over L (in place on Bu -> xs)
  s5_scan_local<<<BSZ * NCHUNK, PDIM, 0, stream>>>(Bu, lamcoef, carry);
  s5_scan_carry<<<BSZ, PDIM, 0, stream>>>(carry, lamcoef);
  s5_scan_fixup<<<BSZ * NCHUNK, PDIM, 0, stream>>>(Bu, lamcoef, carry);

  // 6) GEMM2 + fused D*u + tanh-GELU epilogue
  s5_gemm_wmma<<<MROWS / 32, 256, 0, stream>>>(Bu, W2p, out, u, Dvec, 1);
}
